// AttentionFusionNet2_38397007626358
// MI455X (gfx1250) — compile-verified, matrix operand order checked by reading, not run on hardware
//
#include <hip/hip_runtime.h>
#include <hip/hip_bf16.h>

// ---------------- CDNA5 WMMA types ----------------
typedef __attribute__((ext_vector_type(16))) __bf16 v16bf;
typedef __attribute__((ext_vector_type(8)))  float  v8f;

union Frag {
    v16bf          bf;
    __bf16         h[16];
    unsigned short u[16];
    uint4          q[2];
};

#define BATCH 16
#define SEQ   2048
#define HDIM  512
#define BS    (BATCH * SEQ)   // 32768 rows

__device__ __forceinline__ unsigned short bfbits(float x) {
    __bf16 h = (__bf16)x;                       // native v_cvt -> bf16
    return __builtin_bit_cast(unsigned short, h);
}
__device__ __forceinline__ float bf2f(unsigned short u) {
    return (float)__builtin_bit_cast(__bf16, u);
}
__device__ __forceinline__ float sigmoidf_(float x) { return 1.0f / (1.0f + expf(-x)); }

// ---------------- fp32 -> bf16 bulk conversion (4 elems/thread, packed cvt) ----------------
__global__ void cvt_bf16_kernel(const float* __restrict__ s, unsigned short* __restrict__ d, int n) {
    int i = (blockIdx.x * 256 + threadIdx.x) * 4;
    if (i < n) {
        float4 f = *(const float4*)(s + i);
        uint2 o;
        o.x = (unsigned int)bfbits(f.x) | ((unsigned int)bfbits(f.y) << 16);
        o.y = (unsigned int)bfbits(f.z) | ((unsigned int)bfbits(f.w) << 16);
        *(uint2*)(d + i) = o;
    }
}

__global__ void bias_sum_kernel(const float* __restrict__ a, const float* __restrict__ b,
                                float* __restrict__ o, int n) {
    int i = blockIdx.x * 256 + threadIdx.x;
    if (i < n) o[i] = a[i] + b[i];
}

// ---------------- GEMM: out = act(Abf[M,K] @ Wbf[N,K]^T + bias) ----------------
// block = 256 (8 waves). Wave tile: 64(M) x 16(N)  (4 A-frags x 1 B-frag -> 4 WMMA / K-chunk,
// ~72 live VGPRs: spill-free).  WG tile: 64(M) x 128(N).  grid = (M/64, N/128).
__global__ void __launch_bounds__(256)
gemm_bf16_kernel(const unsigned short* __restrict__ A,
                 const unsigned short* __restrict__ Wbf,
                 const float* __restrict__ bias,
                 float* __restrict__ C, unsigned short* __restrict__ Cbf,
                 int N, int K, int act) {
    const int m0   = blockIdx.x * 64;
    const int wave = threadIdx.x >> 5;
    const int lane = threadIdx.x & 31;
    const int lm   = lane & 15;
    const int kb   = (lane >> 4) * 8;
    const int n0   = blockIdx.y * 128 + wave * 16;

    v8f acc[4] = {};
    const unsigned short* arow[4];
    #pragma unroll
    for (int i = 0; i < 4; ++i) arow[i] = A + (size_t)(m0 + 16*i + lm) * K + kb;
    const unsigned short* wrow = Wbf + (size_t)(n0 + lm) * K + kb;

    for (int k0 = 0; k0 < K; k0 += 32) {
        Frag af[4], bfr;
        #pragma unroll
        for (int i = 0; i < 4; ++i) {
            af[i].q[0] = *(const uint4*)(arow[i] + k0);
            af[i].q[1] = *(const uint4*)(arow[i] + k0 + 16);
        }
        bfr.q[0] = *(const uint4*)(wrow + k0);
        bfr.q[1] = *(const uint4*)(wrow + k0 + 16);
        #pragma unroll
        for (int i = 0; i < 4; ++i)
            acc[i] = __builtin_amdgcn_wmma_f32_16x16x32_bf16(
                         false, af[i].bf, false, bfr.bf, (short)0, acc[i], false, false);
    }

    const int col = n0 + lm;
    const float bv = bias[col];
    #pragma unroll
    for (int i = 0; i < 4; ++i) {
        const int rbase = m0 + 16*i + 8 * (lane >> 4);
        #pragma unroll
        for (int r = 0; r < 8; ++r) {
            float val = acc[i][r] + bv;
            if (act) val = tanhf(val);
            const size_t o = (size_t)(rbase + r) * N + col;
            if (C)   C[o]   = val;
            if (Cbf) Cbf[o] = bfbits(val);
        }
    }
}

// ---------------- concat3 GEMM: ctx = tanh([a|v|l](M,1536) @ Wc(512,1536)^T + bc), bf16 out ----------
__global__ void __launch_bounds__(256)
gemm_concat3_kernel(const unsigned short* __restrict__ Aa,
                    const unsigned short* __restrict__ Av,
                    const unsigned short* __restrict__ Al,
                    const unsigned short* __restrict__ Wbf,
                    const float* __restrict__ bias,
                    unsigned short* __restrict__ Cbf) {
    const int K = 1536, N = 512;
    const int m0   = blockIdx.x * 64;
    const int wave = threadIdx.x >> 5;
    const int lane = threadIdx.x & 31;
    const int lm   = lane & 15;
    const int kb   = (lane >> 4) * 8;
    const int n0   = blockIdx.y * 128 + wave * 16;

    v8f acc[4] = {};
    const unsigned short* wrow = Wbf + (size_t)(n0 + lm) * K + kb;

    for (int k0 = 0; k0 < K; k0 += 32) {
        const int seg = k0 >> 9;                  // 512 % 32 == 0 -> chunks never straddle
        const unsigned short* base = (seg == 0) ? Aa : (seg == 1) ? Av : Al;
        const int ks = (k0 & 511) + kb;
        Frag af[4], bfr;
        #pragma unroll
        for (int i = 0; i < 4; ++i) {
            const unsigned short* arow = base + (size_t)(m0 + 16*i + lm) * 512 + ks;
            af[i].q[0] = *(const uint4*)(arow);
            af[i].q[1] = *(const uint4*)(arow + 16);
        }
        bfr.q[0] = *(const uint4*)(wrow + k0);
        bfr.q[1] = *(const uint4*)(wrow + k0 + 16);
        #pragma unroll
        for (int i = 0; i < 4; ++i)
            acc[i] = __builtin_amdgcn_wmma_f32_16x16x32_bf16(
                         false, af[i].bf, false, bfr.bf, (short)0, acc[i], false, false);
    }

    const int col = n0 + lm;
    const float bv = bias[col];
    #pragma unroll
    for (int i = 0; i < 4; ++i) {
        const int rbase = m0 + 16*i + 8 * (lane >> 4);
        #pragma unroll
        for (int r = 0; r < 8; ++r)
            Cbf[(size_t)(rbase + r) * N + col] = bfbits(tanhf(acc[i][r] + bv));
    }
}

// ---------------- attention fusion: scores -> softmax -> weighted sum (bf16 in/out) -------------
// one wave per row; block 256 => 8 rows/block; grid BS/8
__global__ void __launch_bounds__(256)
fuse_kernel(const unsigned short* __restrict__ a, const unsigned short* __restrict__ v,
            const unsigned short* __restrict__ l, const unsigned short* __restrict__ ctx,
            unsigned short* __restrict__ fus) {
    const int row  = blockIdx.x * 8 + (threadIdx.x >> 5);
    const int lane = threadIdx.x & 31;
    const size_t base = (size_t)row * HDIM;
    float sa = 0.f, sv = 0.f, sl = 0.f;
    float av[HDIM / 32], vv[HDIM / 32], lv[HDIM / 32];
    #pragma unroll
    for (int i = 0; i < HDIM / 32; ++i) {
        int h = lane + 32 * i;
        float cx = bf2f(ctx[base + h]);
        av[i] = bf2f(a[base + h]);
        vv[i] = bf2f(v[base + h]);
        lv[i] = bf2f(l[base + h]);
        sa += av[i] * cx;
        sv += vv[i] * cx;
        sl += lv[i] * cx;
    }
    #pragma unroll
    for (int off = 16; off > 0; off >>= 1) {
        sa += __shfl_xor(sa, off, 32);
        sv += __shfl_xor(sv, off, 32);
        sl += __shfl_xor(sl, off, 32);
    }
    float m  = fmaxf(sa, fmaxf(sv, sl));
    float ea = expf(sa - m), ev = expf(sv - m), el = expf(sl - m);
    float inv = 1.0f / (ea + ev + el);
    float wa = ea * inv, wv = ev * inv, wl = el * inv;
    #pragma unroll
    for (int i = 0; i < HDIM / 32; ++i) {
        int h = lane + 32 * i;
        fus[base + h] = bfbits(wa * av[i] + wv * vv[i] + wl * lv[i]);
    }
}

// ---------------- persistent LSTM recurrence ----------------
// 16 workgroups x 256 threads. WG w owns cols [w*32, w*32+32) of each gate.
// Whh slice (128 gate-rows x 512, bf16) is LDS-resident for all 2048 steps.
__device__ __forceinline__ void grid_barrier(int* bar, int target) {
    __syncthreads();
    if (threadIdx.x == 0) {
        __threadfence();
        atomicAdd(bar, 1);
        while (__hip_atomic_load(bar, __ATOMIC_ACQUIRE, __HIP_MEMORY_SCOPE_AGENT) < target) {}
    }
    __syncthreads();
}

__global__ void __launch_bounds__(256)
lstm_kernel(const float* __restrict__ Gx, const unsigned short* __restrict__ whh_bf,
            const float* __restrict__ h0, const float* __restrict__ c0,
            unsigned short* __restrict__ hbuf,       // [2][16*512] bf16 staging
            float* __restrict__ r_out, float* __restrict__ hT,
            float* __restrict__ cT, int* __restrict__ bar) {
    extern __shared__ char smem[];
    unsigned short* whh_lds = (unsigned short*)smem;                       // 128*512*2 = 128KB
    float* gbuf = (float*)(smem + 128 * 512 * 2);                          // 16*128*4  = 8KB
    float* cbuf = (float*)(smem + 128 * 512 * 2 + 16 * 128 * 4);           // 16*32*4   = 2KB

    const int w    = blockIdx.x;          // 0..15
    const int tid  = threadIdx.x;
    const int wave = tid >> 5;
    const int lane = tid & 31;
    const int lm   = lane & 15;
    const int kb   = (lane >> 4) * 8;

    // Preload this WG's Whh slice: LDS row idx -> global gate-row (idx/32)*512 + w*32 + idx%32
    for (int u = tid; u < (128 * 512) / 8; u += 256) {
        int e = u * 8, idx = e >> 9, k = e & 511;
        int grow = (idx >> 5) * 512 + w * 32 + (idx & 31);
        *(uint4*)(whh_lds + idx * 512 + k) = *(const uint4*)(whh_bf + (size_t)grow * 512 + k);
    }
    // Init c slice (LDS-resident across all steps) and this WG's share of h staging buffer 0
    for (int e = tid; e < 512; e += 256) {
        int b = e >> 5, c = e & 31;
        cbuf[b * 32 + c] = c0[b * HDIM + w * 32 + c];
        hbuf[w * 512 + e] = bfbits(h0[w * 512 + e]);   // flat 8192-elem init, 512 per WG
    }
    __syncthreads();
    grid_barrier(bar, 16);

    const int idx   = wave * 16 + lm;                     // owned output tile column
    const int nglob = (idx >> 5) * 512 + w * 32 + (idx & 31);
    int buf = 0;

    for (int t = 0; t < SEQ; ++t) {
        const unsigned short* h = hbuf + buf * (BATCH * HDIM);
        const unsigned short* hrow = h + lm * HDIM;
        v8f acc = {};
        #pragma unroll 4
        for (int k0 = 0; k0 < HDIM; k0 += 32) {
            Frag af, bfr;
            af.q[0]  = *(const uint4*)(hrow + k0 + kb);
            af.q[1]  = *(const uint4*)(hrow + k0 + kb + 16);
            const unsigned short* wl_ = whh_lds + idx * 512 + k0 + kb;   // ds_load_b128
            bfr.q[0] = *(const uint4*)(wl_);
            bfr.q[1] = *(const uint4*)(wl_ + 16);
            acc = __builtin_amdgcn_wmma_f32_16x16x32_bf16(
                      false, af.bf, false, bfr.bf, (short)0, acc, false, false);
        }
        #pragma unroll
        for (int r = 0; r < 8; ++r) {
            int b = r + 8 * (lane >> 4);
            gbuf[b * 128 + idx] = acc[r] + Gx[((size_t)(b * SEQ + t)) * 2048 + nglob];
        }
        __syncthreads();

        for (int e = tid; e < 512; e += 256) {
            int b = e >> 5, c = e & 31;
            float iv = gbuf[b * 128 + c];
            float fv = gbuf[b * 128 + 32 + c];
            float gv = gbuf[b * 128 + 64 + c];
            float ov = gbuf[b * 128 + 96 + c];
            float cc = sigmoidf_(fv) * cbuf[b * 32 + c] + sigmoidf_(iv) * tanhf(gv);
            float hh = sigmoidf_(ov) * tanhf(cc);
            cbuf[b * 32 + c] = cc;
            int col = w * 32 + c;
            r_out[((size_t)b * SEQ + t) * HDIM + col] = hh;
            hbuf[(buf ^ 1) * (BATCH * HDIM) + b * HDIM + col] = bfbits(hh);
            if (t == SEQ - 1) { hT[b * HDIM + col] = hh; cT[b * HDIM + col] = cc; }
        }
        grid_barrier(bar, 16 * (t + 2));
        buf ^= 1;
    }
}

// ---------------- host launch ----------------
static inline size_t align256(size_t x) { return (x + 255) & ~(size_t)255; }

extern "C" void kernel_launch(void* const* d_in, const int* in_sizes, int n_in,
                              void* d_out, int out_size, void* d_ws, size_t ws_size,
                              hipStream_t stream) {
    (void)in_sizes; (void)n_in; (void)out_size; (void)ws_size;
    const float* a_in  = (const float*)d_in[0];
    const float* v_in  = (const float*)d_in[1];
    const float* l_in  = (const float*)d_in[2];
    const float* h0    = (const float*)d_in[3];
    const float* c0    = (const float*)d_in[4];
    const float* Wa    = (const float*)d_in[5];
    const float* ba    = (const float*)d_in[6];
    const float* Wv    = (const float*)d_in[7];
    const float* bv    = (const float*)d_in[8];
    const float* Wl    = (const float*)d_in[9];
    const float* bl    = (const float*)d_in[10];
    const float* Wc    = (const float*)d_in[11];
    const float* bc    = (const float*)d_in[12];
    const float* Wih   = (const float*)d_in[13];
    const float* Whh   = (const float*)d_in[14];
    const float* bih   = (const float*)d_in[15];
    const float* bhh   = (const float*)d_in[16];

    char* ws = (char*)d_ws;
    size_t cur = 0;
    auto take = [&](size_t bytes) { size_t r = cur; cur = align256(cur + bytes); return r; };

    const size_t F = sizeof(float), U = sizeof(unsigned short);
    unsigned short* ain_bf = (unsigned short*)(ws + take((size_t)BS * 512 * U));
    unsigned short* vin_bf = (unsigned short*)(ws + take((size_t)BS * 768 * U));
    unsigned short* lin_bf = (unsigned short*)(ws + take((size_t)BS * 1024 * U));
    unsigned short* a_bf   = (unsigned short*)(ws + take((size_t)BS * 512 * U));
    unsigned short* v_bf   = (unsigned short*)(ws + take((size_t)BS * 512 * U));
    unsigned short* l_bf   = (unsigned short*)(ws + take((size_t)BS * 512 * U));
    unsigned short* ctx_bf = (unsigned short*)(ws + take((size_t)BS * 512 * U));
    unsigned short* fus_bf = (unsigned short*)(ws + take((size_t)BS * 512 * U));
    float* gx_ws = (float*)(ws + take((size_t)BS * 2048 * F));
    unsigned short* wa_bf  = (unsigned short*)(ws + take((size_t)512 * 512 * U));
    unsigned short* wv_bf  = (unsigned short*)(ws + take((size_t)512 * 768 * U));
    unsigned short* wl_bf  = (unsigned short*)(ws + take((size_t)512 * 1024 * U));
    unsigned short* wc_bf  = (unsigned short*)(ws + take((size_t)512 * 1536 * U));
    unsigned short* wih_bf = (unsigned short*)(ws + take((size_t)2048 * 512 * U));
    unsigned short* whh_bf = (unsigned short*)(ws + take((size_t)2048 * 512 * U));
    float* bsum  = (float*)(ws + take(2048 * F));
    unsigned short* hbuf = (unsigned short*)(ws + take((size_t)2 * BATCH * HDIM * U));
    int* bar = (int*)(ws + take(256));

    hipMemsetAsync(bar, 0, 256, stream);

    auto cvtN = [&](const float* s, unsigned short* d, size_t n) {
        cvt_bf16_kernel<<<(unsigned)((n / 4 + 255) / 256), 256, 0, stream>>>(s, d, (int)n);
    };
    // weights -> bf16
    cvtN(Wa,  wa_bf,  (size_t)512 * 512);
    cvtN(Wv,  wv_bf,  (size_t)512 * 768);
    cvtN(Wl,  wl_bf,  (size_t)512 * 1024);
    cvtN(Wc,  wc_bf,  (size_t)512 * 1536);
    cvtN(Wih, wih_bf, (size_t)2048 * 512);
    cvtN(Whh, whh_bf, (size_t)2048 * 512);
    bias_sum_kernel<<<8, 256, 0, stream>>>(bih, bhh, bsum, 2048);
    // activations -> bf16
    cvtN(a_in, ain_bf, (size_t)BS * 512);
    cvtN(v_in, vin_bf, (size_t)BS * 768);
    cvtN(l_in, lin_bf, (size_t)BS * 1024);

    // projections + context + Gx (all bf16 WMMA GEMMs, 64x128 WG tiles)
    gemm_bf16_kernel<<<dim3(BS / 64, 4), 256, 0, stream>>>(ain_bf, wa_bf, ba, nullptr, a_bf, 512, 512, 0);
    gemm_bf16_kernel<<<dim3(BS / 64, 4), 256, 0, stream>>>(vin_bf, wv_bf, bv, nullptr, v_bf, 512, 768, 0);
    gemm_bf16_kernel<<<dim3(BS / 64, 4), 256, 0, stream>>>(lin_bf, wl_bf, bl, nullptr, l_bf, 512, 1024, 0);
    gemm_concat3_kernel<<<dim3(BS / 64, 4), 256, 0, stream>>>(a_bf, v_bf, l_bf, wc_bf, bc, ctx_bf);
    fuse_kernel<<<BS / 8, 256, 0, stream>>>(a_bf, v_bf, l_bf, ctx_bf, fus_bf);
    gemm_bf16_kernel<<<dim3(BS / 64, 16), 256, 0, stream>>>(fus_bf, wih_bf, bsum, gx_ws, nullptr, 2048, 512, 0);

    // persistent recurrence: 16 WGs, Whh LDS-resident (128KB) + gate/cell buffers
    float* r_out = (float*)d_out;
    float* hT    = r_out + (size_t)BATCH * SEQ * HDIM;
    float* cT    = hT + BATCH * HDIM;
    const size_t smem = 128 * 512 * 2 + 16 * 128 * 4 + 16 * 32 * 4;   // 141312 B
    lstm_kernel<<<16, 256, smem, stream>>>(gx_ws, whh_bf, h0, c0, hbuf, r_out, hT, cT, bar);
}
